// ReconstructionModel_8890582302906
// MI455X (gfx1250) — matrix operand order repaired and checked by reading, not compiled, not run on hardware
//
#include <hip/hip_runtime.h>
#include <hip/hip_bf16.h>

typedef __attribute__((ext_vector_type(16))) _Float16 v16h;
typedef __attribute__((ext_vector_type(8)))  _Float16 h8;
typedef __attribute__((ext_vector_type(8)))  float    v8f;

union U16h { v16h v; h8 h2[2]; _Float16 e[16]; };
union U8f  { v8f  v; float e[8]; };

#define NB    8      // batch
#define TT    2048   // time
#define FD    161    // features
#define SW    48     // SIDE window
#define PADC  12     // max conv pad (25//2)
#define HW    72     // SW + 2*PADC (padded time extent)
#define NWAVE 8      // 256 threads / wave32

constexpr int KS[6]   = {25,21,17,13,9,5};
constexpr int CH[5]   = {161,80,40,20,10};
constexpr int OCP[4]  = {80,48,32,16};          // oc padded to 16
constexpr int ICPW[4] = {192,96,64,32};         // layer-input channels padded to 32
constexpr int DIL[5]  = {1,2,4,8,16};
constexpr int DLEN[6] = {48,47,45,41,33,17};

// K extent per [branch][layer]: KS[bi]*ICPW[li]  (dk-major x padded-channel flattening)
constexpr int KTP[6][4] = {
  {4800,2400,1600,800},   // k=25
  {4032,2016,1344,672},   // k=21
  {3264,1632,1088,544},   // k=17
  {2496,1248, 832,416},   // k=13
  {1728, 864, 576,288},   // k=9
  { 960, 480, 320,160},   // k=5
};
// f16 arena offsets (halves); rows = OCP[li], row stride = KTP[bi][li]
constexpr int WOFF[6][4] = {
  {      0,  384000,  499200,  550400},
  { 563200,  885760,  982528, 1025536},
  {1036288, 1297408, 1375744, 1410560},
  {1419264, 1618944, 1678848, 1705472},
  {1712128, 1850368, 1891840, 1910272},
  {1914880, 1991680, 2014720, 2024960},
};
constexpr int WQ_TOTAL = 2027520;               // halves

// workspace layout (bytes)
constexpr size_t OFF_META  = 0;                 // 16 ints: first[8], last[8]
constexpr size_t OFF_WQ    = 256;               // f16 weight arena
constexpr size_t OFF_STATE = ((OFF_WQ + (size_t)WQ_TOTAL*2 + 4095) / 4096) * 4096;
constexpr size_t STATE_SZ  = (size_t)NB*SW*FD*4;
constexpr size_t OFF_FS    = OFF_STATE + 4*STATE_SZ;   // fsout [4][B][161] f32

struct ModelPtrs {
  const float* x;
  const int*   mask;
  const float* aeW[24]; const float* aeB[24]; const float* aeT[24];
  const float* dnW[2][5]; const float* dnB[2][5]; const float* dnT[2][5];
  const float* lnW[2][2]; const float* lnB[2][2]; const float* lnT[2][2];
  const float* fnW[2]; const float* fnB[2]; const float* fnT[2];
};

__device__ __forceinline__ float taylor5(float x, float t0, float t1, float t2, float t3, float t4) {
  return t0 + x*(t1 + x*(t2 + x*(t3 + x*t4)));
}

// ---------------- meta: find first/last masked frame per batch ----------------
__global__ void meta_kernel(const int* __restrict__ mask, int* __restrict__ meta) {
  int b = threadIdx.x;
  if (b < NB) {
    int f = -1, lst = -1;
    for (int t = 0; t < TT; ++t) {
      if (mask[b*TT + t] != 0) { if (f < 0) f = t; lst = t; }
    }
    if (f < 0) { f = 0; lst = TT - 1; }
    meta[b]      = f;
    meta[NB + b] = lst;
  }
}

// ---------------- init: slice li/ri windows from x, zero d_out ----------------
__global__ void init_kernel(const float* __restrict__ x, const int* __restrict__ meta,
                            float* __restrict__ li0, float* __restrict__ ri0,
                            float* __restrict__ out, int out_n) {
  int stride = gridDim.x * blockDim.x;
  int total1 = NB*SW*FD;
  for (int i = blockIdx.x*blockDim.x + threadIdx.x; i < total1; i += stride) {
    int b = i / (SW*FD);
    int r = (i / FD) % SW;
    int f = i % FD;
    int first = meta[b], last = meta[NB + b];
    int sl = first - SW; sl = sl < 0 ? 0 : (sl > TT-SW ? TT-SW : sl);
    int sr = last + 1;   sr = sr < 0 ? 0 : (sr > TT-SW ? TT-SW : sr);
    li0[i] = x[((size_t)b*TT + sl + r)*FD + f];
    ri0[i] = x[((size_t)b*TT + sr + r)*FD + f];
  }
  for (int i = blockIdx.x*blockDim.x + threadIdx.x; i < out_n; i += stride) out[i] = 0.f;
}

// -------- weight conversion: f32 [oc][ic][k] -> padded f16 [ocP][k*icP] (dk-major) --------
__global__ void wconv_kernel(ModelPtrs mp, _Float16* __restrict__ wq) {
  int seg = blockIdx.y;          // 0..23 = branch*4 + layer
  int bi = seg >> 2, li = seg & 3;
  int oc  = CH[li+1];
  int ic  = CH[li];
  int k   = KS[bi];
  int icp = ICPW[li];
  int ktp = KTP[bi][li];         // == k*icp
  int ocp = OCP[li];
  const float* w = mp.aeW[bi*4 + li];
  _Float16* dst = wq + WOFF[bi][li];
  int n = ocp * ktp;
  for (int i = blockIdx.x*blockDim.x + threadIdx.x; i < n; i += gridDim.x*blockDim.x) {
    int r  = i / ktp;
    int c  = i - r*ktp;
    int dk = c / icp;
    int ii = c - dk*icp;
    dst[i] = (r < oc && ii < ic) ? (_Float16)w[(r*ic + ii)*k + dk] : (_Float16)0.f;
  }
}

// ---------------- fs: one workgroup per (batch, source-window) ----------------
__global__ __launch_bounds__(256, 1)
void fs_kernel(ModelPtrs mp, const _Float16* __restrict__ wq, const int* __restrict__ meta,
               const float* __restrict__ liCur, const float* __restrict__ riCur,
               float* __restrict__ fsout, int l) {
  // activations: time-major, channel-contiguous, zero-padded (t and channel)
  __shared__ __align__(32) _Float16 hX[HW*192];   // layer0 input  (161 -> 192)
  __shared__ __align__(32) _Float16 hP[HW*96];    // layer0 output (80 -> 96)
  __shared__ __align__(32) _Float16 hQ[HW*64];    // layer1 output (40 -> 64)
  __shared__ __align__(32) _Float16 hR[HW*32];    // layer2 output (20 -> 32)
  __shared__ float sum10[10*SW];
  __shared__ float dping[10*SW];
  __shared__ float dpong[10*SW];
  __shared__ float mvec[10];
  __shared__ float v85[85];

  const int tid  = threadIdx.x;
  const int lane = tid & 31;
  const int wave = tid >> 5;
  const int b = blockIdx.x;
  const int s = blockIdx.y;                       // 0=li, 1=left_remain, 2=ri, 3=right_remain
  const int sideIdx = (s == 1 || s == 2) ? 1 : 0; // 0=left params, 1=right params

  for (int i = tid; i < HW*192; i += 256) hX[i] = (_Float16)0.f;
  for (int i = tid; i < HW*96;  i += 256) hP[i] = (_Float16)0.f;
  for (int i = tid; i < HW*64;  i += 256) hQ[i] = (_Float16)0.f;
  for (int i = tid; i < HW*32;  i += 256) hR[i] = (_Float16)0.f;
  for (int i = tid; i < 10*SW;  i += 256) sum10[i] = 0.f;
  __syncthreads();

  // ---- load window [48][161] -> time-major f16 ----
  const float* src;
  if (s == 0)      src = liCur + (size_t)b*SW*FD;
  else if (s == 2) src = riCur + (size_t)b*SW*FD;
  else {
    int first = meta[b], last = meta[NB + b];
    int st = (s == 1) ? (last + 2 + l) : (first - (2 + l));
    st = st < 0 ? 0 : (st > TT-SW ? TT-SW : st);
    src = mp.x + ((size_t)b*TT + st)*FD;
  }
  for (int i = tid; i < SW*FD; i += 256) {
    int t = i / FD, f = i - t*FD;
    hX[(PADC + t)*192 + f] = (_Float16)src[i];
  }
  __syncthreads();

  // ---- autoencoder1: 6 branches x 4 conv layers as WMMA GEMMs ----
  const int Mlane = lane & 15;
  const int hiA = (lane >= 16) ? 8 : 0;     // A / C-D lane-half row offset
  const int hiB = (lane >= 16) ? 16 : 0;    // B lane-half K offset

  for (int bi = 0; bi < 6; ++bi) {
    const int k = KS[bi];
    const int pad = k >> 1;
    for (int li = 0; li < 4; ++li) {
      const int oc   = CH[li+1];
      const int icp  = ICPW[li];
      const int ktp  = KTP[bi][li];          // k*icp
      const int ocpw = (li < 3) ? ICPW[li+1] : 0;   // output buffer channel stride
      const _Float16* W = wq + WOFF[bi][li]; // [ocP][ktp] f16, zero-padded
      const float* bias = mp.aeB[bi*4 + li];
      const float* tc   = mp.aeT[bi*4 + li];
      const float t0 = tc[0], t1 = tc[1], t2 = tc[2], t3 = tc[3], t4 = tc[4];
      const _Float16* inb = (li == 0) ? hX : ((li == 1) ? hP : ((li == 2) ? hQ : hR));
      _Float16* outb      = (li == 0) ? hP : ((li == 1) ? hQ : hR);

      const int mtiles = OCP[li] >> 4;
      const int tiles  = mtiles * 3;          // 3 N-tiles cover L=48
      const int nkc    = ktp >> 5;

      U8f acc[2];
      #pragma unroll
      for (int r = 0; r < 8; ++r) { acc[0].e[r] = 0.f; acc[1].e[r] = 0.f; }

      for (int slot = 0; slot < 2; ++slot) {
        int tt = wave + slot*NWAVE;
        if (tt >= tiles) continue;
        int mt = tt / 3, nt = tt - mt*3;
        int Mrow = mt*16 + Mlane;
        int Ncol = nt*16 + Mlane;
        // B-tile is a contiguous LDS window: addr(K) = Ncst + K  (dk-major im2col)
        const _Float16* bwin = inb + (Ncol - pad + PADC)*icp + hiB;
        const _Float16* arow = W + (size_t)Mrow*ktp + hiB;
        for (int kc = 0; kc < nkc; ++kc) {
          int kb = kc << 5;
          __builtin_prefetch((const void*)(arow + kb + 32), 0, 1);
          U16h a, bf;
          a.h2[0]  = *(const h8*)(arow + kb);        // K = kb+hiB   .. +7
          a.h2[1]  = *(const h8*)(arow + kb + 8);    // K = kb+hiB+8 .. +15
          bf.h2[0] = *(const h8*)(bwin + kb);        // K = kb+hiB   .. +7
          bf.h2[1] = *(const h8*)(bwin + kb + 8);    // K = kb+hiB+8 .. +15
          acc[slot].v = __builtin_amdgcn_wmma_f32_16x16x32_f16(
              false, a.v, false, bf.v, (short)0, acc[slot].v, false, false);
        }
      }
      __syncthreads();   // all reads of inb done before outb writes
      for (int slot = 0; slot < 2; ++slot) {
        int tt = wave + slot*NWAVE;
        if (tt >= tiles) continue;
        int mt = tt / 3, nt = tt - mt*3;
        int N = nt*16 + Mlane;
        #pragma unroll
        for (int r = 0; r < 8; ++r) {
          int M = mt*16 + r + hiA;
          if (M < oc) {
            float y = acc[slot].e[r] + bias[M];
            y = taylor5(y, t0, t1, t2, t3, t4);
            if (li < 3) outb[(PADC + N)*ocpw + M] = (_Float16)y;
            else        sum10[M*SW + N] += y;   // branch sum (same owner wave per tile)
          }
        }
      }
      __syncthreads();
    }
  }

  // ---- downscale_time_conv: 5 dilated k=2 convs, 10->10, L 48 -> 17 ----
  float* cur = sum10;
  float* nxt = dping;
  for (int ci = 0; ci < 5; ++ci) {
    const float* w   = mp.dnW[sideIdx][ci];
    const float* bb2 = mp.dnB[sideIdx][ci];
    const float* tc  = mp.dnT[sideIdx][ci];
    float t0 = tc[0], t1 = tc[1], t2 = tc[2], t3 = tc[3], t4 = tc[4];
    int dil = DIL[ci];
    int Lo  = DLEN[ci+1];
    int tot = 10 * Lo;
    for (int i = tid; i < tot; i += 256) {
      int ocd = i / Lo, tp = i - ocd*Lo;
      float y = bb2[ocd];
      #pragma unroll
      for (int icd = 0; icd < 10; ++icd) {
        y += w[(ocd*10 + icd)*2 + 0] * cur[icd*SW + tp]
           + w[(ocd*10 + icd)*2 + 1] * cur[icd*SW + tp + dil];
      }
      nxt[ocd*SW + tp] = taylor5(y, t0, t1, t2, t3, t4);
    }
    __syncthreads();
    float* old = cur;
    cur = nxt;
    nxt = (ci == 0) ? dpong : old;
  }

  // ---- mean over t (L=17) ----
  if (tid < 10) {
    float sum = 0.f;
    for (int t = 0; t < 17; ++t) sum += cur[tid*SW + t];
    mvec[tid] = sum * (1.f / 17.f);
  }
  __syncthreads();

  // ---- lin: 10 -> 85 -> 161 ----
  {
    const float* w = mp.lnW[sideIdx][0]; const float* bb2 = mp.lnB[sideIdx][0];
    const float* tc = mp.lnT[sideIdx][0];
    if (tid < 85) {
      float y = bb2[tid];
      #pragma unroll
      for (int i = 0; i < 10; ++i) y += w[tid*10 + i] * mvec[i];
      v85[tid] = taylor5(y, tc[0], tc[1], tc[2], tc[3], tc[4]);
    }
  }
  __syncthreads();
  {
    const float* w = mp.lnW[sideIdx][1]; const float* bb2 = mp.lnB[sideIdx][1];
    const float* tc = mp.lnT[sideIdx][1];
    if (tid < FD) {
      float y = bb2[tid];
      for (int i = 0; i < 85; ++i) y += w[tid*85 + i] * v85[i];
      fsout[((size_t)s*NB + b)*FD + tid] = taylor5(y, tc[0], tc[1], tc[2], tc[3], tc[4]);
    }
  }
}

// ---------------- final MLP + output scatter + autoregressive state shift ----------------
__global__ __launch_bounds__(256, 1)
void final_kernel(ModelPtrs mp, const int* __restrict__ meta, const float* __restrict__ fsout,
                  const float* __restrict__ liCur, const float* __restrict__ riCur,
                  float* __restrict__ liNxt, float* __restrict__ riNxt,
                  float* __restrict__ out, int l, int maxm) {
  __shared__ float z[322];
  __shared__ float hmid[241];
  __shared__ float lf[FD];
  __shared__ float rf[FD];
  int tid = threadIdx.x;
  int b = blockIdx.x;

  for (int pass = 0; pass < 2; ++pass) {
    int s0 = (pass == 0) ? 0 : 2;
    if (tid < 322) {
      int part = (tid < FD) ? 0 : 1;
      int j    = (tid < FD) ? tid : tid - FD;
      z[tid] = fsout[((size_t)(s0 + part)*NB + b)*FD + j];
    }
    __syncthreads();
    {
      const float* w = mp.fnW[0]; const float* bb = mp.fnB[0]; const float* tc = mp.fnT[0];
      if (tid < 241) {
        float y = bb[tid];
        for (int i = 0; i < 322; ++i) y += w[tid*322 + i] * z[i];
        hmid[tid] = taylor5(y, tc[0], tc[1], tc[2], tc[3], tc[4]);
      }
    }
    __syncthreads();
    {
      const float* w = mp.fnW[1]; const float* bb = mp.fnB[1]; const float* tc = mp.fnT[1];
      if (tid < FD) {
        float y = bb[tid];
        for (int i = 0; i < 241; ++i) y += w[tid*241 + i] * hmid[i];
        y = taylor5(y, tc[0], tc[1], tc[2], tc[3], tc[4]);
        y = fminf(fmaxf(y, 0.f), 6.f);
        if (pass == 0) lf[tid] = y; else rf[tid] = y;
      }
    }
    __syncthreads();
  }

  int first = meta[b], last = meta[NB + b];
  int ridx = last - first - l;
  int cap  = maxm - 1 - l;
  if (cap < ridx) ridx = cap;
  bool cond = (l < ridx);

  if (tid < FD && cond) {
    out[((size_t)b*maxm + l)*FD + tid] = lf[tid];      // left write first,
    out[((size_t)b*maxm + ridx)*FD + tid] = rf[tid];   // right overwrites on collision
  }

  for (int i = tid; i < SW*FD; i += 256) {
    int r = i / FD, f = i - r*FD;
    float lv, rv;
    if (cond) {
      lv = (r < SW-1) ? liCur[((size_t)b*SW + r + 1)*FD + f] : lf[f];
      rv = (r > 0)    ? riCur[((size_t)b*SW + r - 1)*FD + f] : rf[f];
    } else {
      lv = liCur[((size_t)b*SW + r)*FD + f];
      rv = riCur[((size_t)b*SW + r)*FD + f];
    }
    liNxt[((size_t)b*SW + r)*FD + f] = lv;
    riNxt[((size_t)b*SW + r)*FD + f] = rv;
  }
}

extern "C" void kernel_launch(void* const* d_in, const int* in_sizes, int n_in,
                              void* d_out, int out_size, void* d_ws, size_t ws_size,
                              hipStream_t stream) {
  (void)in_sizes; (void)n_in; (void)ws_size;

  // jax pytree order: dict keys sorted -> 'ae1', 'down', 'final', 'lin'
  ModelPtrs mp;
  mp.x    = (const float*)d_in[0];
  mp.mask = (const int*)d_in[1];
  int p = 2;
  for (int i = 0; i < 24; ++i) {
    mp.aeW[i] = (const float*)d_in[p];
    mp.aeB[i] = (const float*)d_in[p+1];
    mp.aeT[i] = (const float*)d_in[p+2];
    p += 3;
  }
  for (int s = 0; s < 2; ++s)
    for (int c = 0; c < 5; ++c) {
      mp.dnW[s][c] = (const float*)d_in[p];
      mp.dnB[s][c] = (const float*)d_in[p+1];
      mp.dnT[s][c] = (const float*)d_in[p+2];
      p += 3;
    }
  for (int f = 0; f < 2; ++f) {
    mp.fnW[f] = (const float*)d_in[p];
    mp.fnB[f] = (const float*)d_in[p+1];
    mp.fnT[f] = (const float*)d_in[p+2];
    p += 3;
  }
  for (int s = 0; s < 2; ++s)
    for (int j = 0; j < 2; ++j) {
      mp.lnW[s][j] = (const float*)d_in[p];
      mp.lnB[s][j] = (const float*)d_in[p+1];
      mp.lnT[s][j] = (const float*)d_in[p+2];
      p += 3;
    }

  char* ws = (char*)d_ws;
  int*      meta  = (int*)(ws + OFF_META);
  _Float16* wq    = (_Float16*)(ws + OFF_WQ);
  float* st[4];   // st[0],st[1]=li ping-pong; st[2],st[3]=ri ping-pong
  for (int i = 0; i < 4; ++i) st[i] = (float*)(ws + OFF_STATE + (size_t)i*STATE_SZ);
  float* fsout = (float*)(ws + OFF_FS);
  float* out   = (float*)d_out;

  int maxm  = out_size / (NB * FD);   // 16 for the reference mask
  int steps = maxm / 2;

  meta_kernel<<<1, 64, 0, stream>>>(mp.mask, meta);
  init_kernel<<<128, 256, 0, stream>>>(mp.x, meta, st[0], st[2], out, out_size);
  wconv_kernel<<<dim3(64, 24), 256, 0, stream>>>(mp, wq);

  for (int l = 0; l < steps; ++l) {
    int cur = l & 1;
    fs_kernel<<<dim3(NB, 4), 256, 0, stream>>>(mp, wq, meta, st[cur], st[2+cur], fsout, l);
    final_kernel<<<NB, 256, 0, stream>>>(mp, meta, fsout, st[cur], st[2+cur],
                                         st[1-cur], st[3-cur], out, l, maxm);
  }
}
